// NNConvNet_3770981286443
// MI455X (gfx1250) — matrix operand order repaired and checked by reading, not compiled, not run on hardware
//
#include <hip/hip_runtime.h>
#include <hip/hip_bf16.h>
#include <cstdint>

typedef __bf16 v16bf  __attribute__((ext_vector_type(16)));
typedef __bf16 bf16x8 __attribute__((ext_vector_type(8)));
typedef float  v8f    __attribute__((ext_vector_type(8)));

#define NNODE 10000
#define NEDGE 160000

__device__ __forceinline__ float silu_f(float z) {
    return z / (1.0f + __expf(-z));
}

// ---------------------------------------------------------------------------
// Edge degree counts (shared by all layers)
// ---------------------------------------------------------------------------
__global__ void count_edges(const int* __restrict__ dst, float* __restrict__ cnt, int E) {
    int i = blockIdx.x * blockDim.x + threadIdx.x;
    if (i < E) atomicAdd(&cnt[dst[i]], 1.0f);
}

// ---------------------------------------------------------------------------
// Stage 1: t[e,j] = silu(edge_attr[e,:] @ w1[:,j] + b1[j])  -> bf16 [E,m]
// 8 outputs per thread, one b128 store.
// ---------------------------------------------------------------------------
__global__ void stage1(const float* __restrict__ ea, const float* __restrict__ w1,
                       const float* __restrict__ b1, __bf16* __restrict__ t,
                       int m, int total8) {
    int idx = blockIdx.x * blockDim.x + threadIdx.x;
    if (idx >= total8) return;
    const int nj = m >> 3;
    const int j0 = (idx % nj) << 3;
    const int e  = idx / nj;
    const float a0 = ea[3 * e + 0], a1 = ea[3 * e + 1], a2 = ea[3 * e + 2];
    bf16x8 out;
#pragma unroll
    for (int u = 0; u < 8; ++u) {
        int j = j0 + u;
        float z = fmaf(a0, w1[j], fmaf(a1, w1[m + j], fmaf(a2, w1[2 * m + j], b1[j])));
        out[u] = (__bf16)silu_f(z);
    }
    *reinterpret_cast<bf16x8*>(t + (size_t)e * m + j0) = out;
}

// ---------------------------------------------------------------------------
// Pack w2 [m,m] f32 row-major into WMMA B-fragment order (bf16).
// Layout: out[((kb*(m/16)+cb)*32 + lane)*16 + j]
//   lane<16 : B[kb*32 + (j<8 ? j : j+8)     , cb*16 + lane]      (K=0..7,16..23)
//   lane>=16: B[kb*32 + 8 + (j<8 ? j : j+8) , cb*16 + (lane-16)] (K=8..15,24..31)
// ---------------------------------------------------------------------------
__global__ void pack_b(const float* __restrict__ w2, __bf16* __restrict__ out, int m) {
    int idx = blockIdx.x * blockDim.x + threadIdx.x;
    if (idx >= m * m) return;
    int j    = idx & 15;
    int lane = (idx >> 4) & 31;
    int cb   = (idx >> 9) % (m / 16);
    int kb   = (idx >> 9) / (m / 16);
    int k    = kb * 32 + ((lane >> 4) << 3) + j + ((j >> 3) << 3);
    int col  = cb * 16 + (lane & 15);
    out[idx] = (__bf16)w2[k * m + col];
}

// ---------------------------------------------------------------------------
// Fused: we = t @ w2 + b2  (bf16 WMMA, f32 accum), then per-edge matvec
// msg[e,o] = sum_i x[src[e],i] * we[e,i*DOUT+o], scatter-add into nsum[dst[e]].
// 8 waves/block: WR row-groups x WC col-waves, RPT row tiles per wave.
// Since COLBLOCK % DOUT == 0, each wave's output column o is invariant across
// the col-block loop -> accumulate msg in registers, commit LDS atomics once.
// ---------------------------------------------------------------------------
template <int DIN, int DOUT, int M_, int WR, int WC, int RPT>
__global__ __launch_bounds__(256) void edge_gemm_fused(
    const __bf16* __restrict__ t,    // [E, M_] bf16
    const __bf16* __restrict__ w2p,  // packed B fragments
    const float*  __restrict__ b2,   // [M_]
    const float*  __restrict__ x,    // [N, DIN]
    const int*    __restrict__ src,  // [E]
    const int*    __restrict__ dst,  // [E]
    float*        __restrict__ nsum) // [N, DOUT]
{
    constexpr int ROWTILE  = WR * RPT * 16;
    constexpr int COLBLOCK = WC * 16;
    static_assert(COLBLOCK % DOUT == 0, "o must be cb-invariant");
    static_assert(WR * WC == 8, "8 waves per block");
    __shared__ float xg[ROWTILE][DIN];
    __shared__ float msg[ROWTILE][DOUT];
    __shared__ int   dsts[ROWTILE];

    const int tid = threadIdx.x;
    const int e0  = blockIdx.x * ROWTILE;

    for (int idx = tid; idx < ROWTILE * DIN; idx += 256) {
        int r = idx / DIN, i = idx % DIN;
        xg[r][i] = x[(size_t)src[e0 + r] * DIN + i];
    }
    for (int idx = tid; idx < ROWTILE * DOUT; idx += 256)
        msg[idx / DOUT][idx % DOUT] = 0.0f;
    if (tid < ROWTILE) dsts[tid] = dst[e0 + tid];
    __syncthreads();

    const int lane = tid & 31;
    const int wave = tid >> 5;
    const int wr   = wave / WC;
    const int wc   = wave % WC;
    const int koff = (lane >> 4) << 3;  // ISA 16-bit A layout: lanes 16-31 hold K+8
    const v16bf* bptr = reinterpret_cast<const v16bf*>(w2p);

    const __bf16* aptr[RPT];
#pragma unroll
    for (int p = 0; p < RPT; ++p) {
        int row = (wr * RPT + p) * 16 + (lane & 15);
        aptr[p] = t + (size_t)(e0 + row) * M_ + koff;
    }

    const int col_lo = wc * 16 + (lane & 15);  // low part of this lane's column
    const int o      = col_lo % DOUT;          // invariant across cb
    v8f macc[RPT] = {};                        // register msg accumulators

    for (int cb = 0; cb < M_; cb += COLBLOCK) {
        const int col0 = cb + wc * 16;
        v8f acc[RPT] = {};
#pragma unroll 2
        for (int kb = 0; kb < M_; kb += 32) {
            v16bf b = bptr[((kb >> 5) * (M_ / 16) + (col0 >> 4)) * 32 + lane];
#pragma unroll
            for (int p = 0; p < RPT; ++p) {
                bf16x8 alo = *reinterpret_cast<const bf16x8*>(aptr[p] + kb);
                bf16x8 ahi = *reinterpret_cast<const bf16x8*>(aptr[p] + kb + 16);
                v16bf a = __builtin_shufflevector(alo, ahi,
                    0,1,2,3,4,5,6,7,8,9,10,11,12,13,14,15);
                acc[p] = __builtin_amdgcn_wmma_f32_16x16x32_bf16(
                    false, a, false, b, (short)0, acc[p], false, false);
            }
        }
        // epilogue: we = acc + b2[col]; macc += xg[:,i] * we   (pure fma + ds_load)
        const int   col = cb + col_lo;
        const int   i   = col / DOUT;
        const float bv  = b2[col];
#pragma unroll
        for (int p = 0; p < RPT; ++p) {
            const int rbase = (wr * RPT + p) * 16 + ((lane >> 4) << 3);
#pragma unroll
            for (int v = 0; v < 8; ++v)
                macc[p][v] = fmaf(acc[p][v] + bv, xg[rbase + v][i], macc[p][v]);
        }
    }

    // commit register accumulators to LDS msg (one atomic batch per wave)
#pragma unroll
    for (int p = 0; p < RPT; ++p) {
        const int rbase = (wr * RPT + p) * 16 + ((lane >> 4) << 3);
#pragma unroll
        for (int v = 0; v < 8; ++v)
            atomicAdd(&msg[rbase + v][o], macc[p][v]);
    }
    __syncthreads();

    for (int idx = tid; idx < ROWTILE * DOUT; idx += 256) {
        int r = idx / DOUT, oo = idx % DOUT;
        atomicAdd(&nsum[(size_t)dsts[r] * DOUT + oo], msg[r][oo]);
    }
}

// ---------------------------------------------------------------------------
// out[n,o] = nsum[n,o]/max(cnt,1) + (x @ root)[n,o] + bias[o]  (+silu)
// ---------------------------------------------------------------------------
template <int DIN, int DOUT, bool ACT>
__global__ void finalize(const float* __restrict__ nsum, const float* __restrict__ cnt,
                         const float* __restrict__ x, const float* __restrict__ root,
                         const float* __restrict__ bias, float* __restrict__ out, int N) {
    int idx = blockIdx.x * blockDim.x + threadIdx.x;
    if (idx >= N * DOUT) return;
    int n = idx / DOUT, o = idx % DOUT;
    float c = cnt[n];
    c = c > 1.0f ? c : 1.0f;
    float acc = nsum[idx] / c + bias[o];
#pragma unroll
    for (int i = 0; i < DIN; ++i) acc = fmaf(x[n * DIN + i], root[i * DOUT + o], acc);
    if (ACT) acc = silu_f(acc);
    out[idx] = acc;
}

// ---------------------------------------------------------------------------
static inline size_t al256(size_t v) { return (v + 255) & ~(size_t)255; }

extern "C" void kernel_launch(void* const* d_in, const int* in_sizes, int n_in,
                              void* d_out, int out_size, void* d_ws, size_t ws_size,
                              hipStream_t stream) {
    (void)in_sizes; (void)n_in; (void)out_size; (void)ws_size;
    const float* x  = (const float*)d_in[0];
    const int*   ei = (const int*)d_in[1];
    const float* ea = (const float*)d_in[2];
    const float* w1[3] = {(const float*)d_in[3], (const float*)d_in[9],  (const float*)d_in[15]};
    const float* b1[3] = {(const float*)d_in[4], (const float*)d_in[10], (const float*)d_in[16]};
    const float* w2[3] = {(const float*)d_in[5], (const float*)d_in[11], (const float*)d_in[17]};
    const float* b2[3] = {(const float*)d_in[6], (const float*)d_in[12], (const float*)d_in[18]};
    const float* rt[3] = {(const float*)d_in[7], (const float*)d_in[13], (const float*)d_in[19]};
    const float* bs[3] = {(const float*)d_in[8], (const float*)d_in[14], (const float*)d_in[20]};
    const int* src = ei;
    const int* dst = ei + NEDGE;

    // workspace carve-up
    char*  ws   = (char*)d_ws;
    size_t off  = 0;
    __bf16* t    = (__bf16*)(ws + off); off += al256((size_t)NEDGE * 1024 * sizeof(__bf16));
    __bf16* w2p  = (__bf16*)(ws + off); off += al256((size_t)1024 * 1024 * sizeof(__bf16));
    float*  nsum = (float*) (ws + off); off += al256((size_t)NNODE * 32 * sizeof(float));
    float*  cnt  = (float*) (ws + off); off += al256((size_t)NNODE * sizeof(float));
    float*  h1   = (float*) (ws + off); off += al256((size_t)NNODE * 32 * sizeof(float));
    float*  h2   = (float*) (ws + off); off += al256((size_t)NNODE * 32 * sizeof(float));

    hipMemsetAsync(cnt, 0, (size_t)NNODE * sizeof(float), stream);
    count_edges<<<(NEDGE + 255) / 256, 256, 0, stream>>>(dst, cnt, NEDGE);

    // ---- Layer 0: din=4, dout=32, m=128 ----
    {
        const int m = 128, tot8 = NEDGE * (m / 8);
        stage1<<<(tot8 + 255) / 256, 256, 0, stream>>>(ea, w1[0], b1[0], t, m, tot8);
        pack_b<<<(m * m + 255) / 256, 256, 0, stream>>>(w2[0], w2p, m);
        hipMemsetAsync(nsum, 0, (size_t)NNODE * 32 * sizeof(float), stream);
        edge_gemm_fused<4, 32, 128, 2, 4, 2><<<NEDGE / 64, 256, 0, stream>>>(
            t, w2p, b2[0], x, src, dst, nsum);
        finalize<4, 32, true><<<(NNODE * 32 + 255) / 256, 256, 0, stream>>>(
            nsum, cnt, x, rt[0], bs[0], h1, NNODE);
    }
    // ---- Layer 1: din=32, dout=32, m=1024 (the 335 GFLOP GEMM) ----
    {
        const int m = 1024, tot8 = NEDGE * (m / 8);
        stage1<<<(tot8 + 255) / 256, 256, 0, stream>>>(ea, w1[1], b1[1], t, m, tot8);
        pack_b<<<(m * m + 255) / 256, 256, 0, stream>>>(w2[1], w2p, m);
        hipMemsetAsync(nsum, 0, (size_t)NNODE * 32 * sizeof(float), stream);
        edge_gemm_fused<32, 32, 1024, 2, 4, 4><<<NEDGE / 128, 256, 0, stream>>>(
            t, w2p, b2[1], h1, src, dst, nsum);
        finalize<32, 32, true><<<(NNODE * 32 + 255) / 256, 256, 0, stream>>>(
            nsum, cnt, h1, rt[1], bs[1], h2, NNODE);
    }
    // ---- Layer 2: din=32, dout=1, m=32 ----
    {
        const int m = 32, tot8 = NEDGE * (m / 8);
        stage1<<<(tot8 + 255) / 256, 256, 0, stream>>>(ea, w1[2], b1[2], t, m, tot8);
        pack_b<<<(m * m + 255) / 256, 256, 0, stream>>>(w2[2], w2p, m);
        hipMemsetAsync(nsum, 0, (size_t)NNODE * 32 * sizeof(float), stream);
        edge_gemm_fused<32, 1, 32, 4, 2, 1><<<NEDGE / 64, 256, 0, stream>>>(
            t, w2p, b2[2], h2, src, dst, nsum);
        finalize<32, 1, false><<<(NNODE + 255) / 256, 256, 0, stream>>>(
            nsum, cnt, h2, rt[2], bs[2], (float*)d_out, NNODE);
    }
}
